// Encoder_6270652252195
// MI455X (gfx1250) — matrix operand order repaired and checked by reading, not compiled, not run on hardware
//
#include <hip/hip_runtime.h>

typedef __attribute__((ext_vector_type(16))) _Float16 v16h;
typedef __attribute__((ext_vector_type(8)))  _Float16 v8h;
typedef __attribute__((ext_vector_type(8)))  float    v8f;
typedef __attribute__((ext_vector_type(4)))  int      v4i;

constexpr int Bn = 64;     // batch
constexpr int Tn = 512;    // time steps
constexpr int Dn = 256;    // input dim
constexpr int Un = 256;    // GRU units
constexpr int G3 = 3 * Un; // 768 (3 gates)

#if __has_builtin(__builtin_amdgcn_global_load_async_to_lds_b128)
#define HAVE_ASYNC_LDS 1
#endif

__device__ __forceinline__ void wait_asynccnt0() {
#if __has_builtin(__builtin_amdgcn_s_wait_asynccnt)
  __builtin_amdgcn_s_wait_asynccnt(0);
#else
  asm volatile("s_wait_asynccnt 0x0" ::: "memory");
#endif
}

// ---------------------------------------------------------------------------
// WMMA helper: D = A(16x32 f16) * B(32x16 f16) + C(16x16 f32)
// ---------------------------------------------------------------------------
__device__ __forceinline__ v8f wmma16(v16h a, v16h b, v8f c) {
  return __builtin_amdgcn_wmma_f32_16x16x32_f16(
      /*neg_a=*/false, a, /*neg_b=*/false, b,
      /*c_mod=*/(short)0, c, /*reuse_a=*/false, /*reuse_b=*/false);
}

// A fragment (16x32) from row-major f16, base = &A[mTile*16*lda + kTile*32].
// Lanes 0-15: m=lane, K = {0..7, 16..23}; lanes 16-31: K = {8..15, 24..31}.
__device__ __forceinline__ v16h load_a_h(const _Float16* base, int lda, int lane) {
  const int m  = lane & 15;
  const int kb = (lane >> 4) << 3;
  const _Float16* p = base + (size_t)m * lda + kb;
  v8h lo = *(const v8h*)(p);
  v8h hi = *(const v8h*)(p + 16);
  v16h a;
#pragma unroll
  for (int i = 0; i < 8; ++i) { a[i] = lo[i]; a[i + 8] = hi[i]; }
  return a;
}

// B fragment (32x16) from pre-transposed weights WT[N][K] (row-major, f16).
// base = &WT[nTile*16 * ldb + kTile*32]; lane n=lane%16, K = {0..7,16..23} (lanes 0-15) / {8..15,24..31} (lanes 16-31).
__device__ __forceinline__ v16h load_b_h(const _Float16* base, int ldb, int lane) {
  const int n  = lane & 15;
  const int kb = (lane >> 4) << 3;
  const _Float16* p = base + (size_t)n * ldb + kb;
  v8h lo = *(const v8h*)(p);
  v8h hi = *(const v8h*)(p + 16);
  v16h b;
#pragma unroll
  for (int i = 0; i < 8; ++i) { b[i] = lo[i]; b[i + 8] = hi[i]; }
  return b;
}

__device__ __forceinline__ float sigmf(float x) {
  return 1.0f / (1.0f + __expf(-x));
}

// ---------------------------------------------------------------------------
// Elementwise f32 -> f16 convert
// ---------------------------------------------------------------------------
__global__ void k_cvt_f16(const float* __restrict__ src, _Float16* __restrict__ dst, int n) {
  int i = blockIdx.x * blockDim.x + threadIdx.x;
  if (i < n) dst[i] = (_Float16)src[i];
}

// Transpose-convert: src[K][N] f32 -> dst[N][K] f16
__global__ void k_transpose_f16(const float* __restrict__ src, _Float16* __restrict__ dst,
                                int K, int N) {
  int i = blockIdx.x * blockDim.x + threadIdx.x;
  if (i < K * N) {
    int k = i / N, n = i % N;
    dst[(size_t)n * K + k] = (_Float16)src[i];
  }
}

// ---------------------------------------------------------------------------
// GEMM: C[M,N] (f16) = A[M,K](f16) @ BT[N,K]^T + bias[N] (f32)
// Block = 256 threads = 8 waves; wave w -> m-tiles 2*(blockIdx.y*8+w)+{0,1},
// n-tiles blockIdx.x*4 .. +3 (B fragments reused across the 2 m-tiles).
// M % 256 == 0, N % 64 == 0, K % 32 == 0.
// ---------------------------------------------------------------------------
__global__ __launch_bounds__(256) void k_gemm_bias(
    const _Float16* __restrict__ A, const _Float16* __restrict__ BT,
    const float* __restrict__ bias, _Float16* __restrict__ C,
    int M, int N, int K) {
  const int lane = threadIdx.x & 31;
  const int wave = threadIdx.x >> 5;
  const int mT0  = (blockIdx.y * 8 + wave) * 2;
  const int nT0  = blockIdx.x * 4;
  const _Float16* A0 = A + (size_t)mT0 * 16 * K;
  const _Float16* A1 = A0 + (size_t)16 * K;

  v8f acc[2][4] = {};
  for (int kt = 0; kt < K; kt += 32) {
    if (kt + 32 < K) {
      __builtin_prefetch(A0 + (size_t)(lane & 15) * K + kt + 32, 0, 3);
      __builtin_prefetch(A1 + (size_t)(lane & 15) * K + kt + 32, 0, 3);
    }
    v16h a0 = load_a_h(A0 + kt, K, lane);
    v16h a1 = load_a_h(A1 + kt, K, lane);
#pragma unroll
    for (int j = 0; j < 4; ++j) {
      v16h b = load_b_h(BT + (size_t)(nT0 + j) * 16 * K + kt, K, lane);
      acc[0][j] = wmma16(a0, b, acc[0][j]);
      acc[1][j] = wmma16(a1, b, acc[1][j]);
    }
  }
  const int nloc = lane & 15;
  const int mhi  = (lane >> 4) << 3;
#pragma unroll
  for (int t = 0; t < 2; ++t) {
#pragma unroll
    for (int j = 0; j < 4; ++j) {
      const int n  = (nT0 + j) * 16 + nloc;
      const float bv = bias[n];
#pragma unroll
      for (int r = 0; r < 8; ++r) {
        const int m = (mT0 + t) * 16 + mhi + r;
        C[(size_t)m * N + n] = (_Float16)(acc[t][j][r] + bv);
      }
    }
  }
}

// ---------------------------------------------------------------------------
// Persistent GRU scan. One 512-thread workgroup per direction (gridDim.x=2).
// h lives in LDS: f32 master state (recurrence accuracy) + f16 shadow (matvec).
// Wave w (= gate-column tile j, 0..15) computes gh tiles for all 4 m-tiles at
// n = {j, 16+j, 32+j}*16 -> each recurrent-weight fragment is loaded from L2
// exactly once per step (384 KB minimum), reused across 4 WMMAs, and all three
// gates for a given (b,u) are lane-local for the sigmoid/tanh epilogue.
// ---------------------------------------------------------------------------
__global__ __launch_bounds__(512) void k_gru_scan(
    const _Float16* __restrict__ gx_f, const _Float16* __restrict__ gx_b, // [B*T, 768]
    const _Float16* __restrict__ Wu_f, const _Float16* __restrict__ Wu_b, // [768, 256] (N-major)
    const float* __restrict__ br_f, const float* __restrict__ br_b,      // [768]
    const float* __restrict__ h0,                                        // [64, 256]
    _Float16* __restrict__ seq_out, int seq_col_f, int seq_col_b,        // pred16 or null
    _Float16* __restrict__ hid_out, int hid_col_f, int hid_col_b) {      // [64,512] or null
  const int dir = blockIdx.x;
  const _Float16* gx = dir ? gx_b : gx_f;
  const _Float16* Wu = dir ? Wu_b : Wu_f;
  const float* br    = dir ? br_b : br_f;
  const int seq_col  = dir ? seq_col_b : seq_col_f;
  const int hid_col  = dir ? hid_col_b : hid_col_f;

  __shared__ __align__(16) float    h32[Bn * Un];  // 64 KB master state
  __shared__ __align__(16) _Float16 h16[Bn * Un];  // 32 KB matvec copy

#ifdef HAVE_ASYNC_LDS
  {
    typedef __attribute__((address_space(1))) v4i* gp_t;
    typedef __attribute__((address_space(3))) v4i* lp_t;
    gp_t gsrc = (gp_t)h0;
    lp_t ldst = (lp_t)h32;
    const int nvec = Bn * Un * 4 / 16;  // 16-byte packets
    for (int i = threadIdx.x; i < nvec; i += (int)blockDim.x)
      __builtin_amdgcn_global_load_async_to_lds_b128(gsrc + i, ldst + i, 0, 0);
    wait_asynccnt0();
  }
  __syncthreads();
  for (int i = threadIdx.x; i < Bn * Un; i += blockDim.x) h16[i] = (_Float16)h32[i];
#else
  for (int i = threadIdx.x; i < Bn * Un; i += blockDim.x) {
    float v = h0[i];
    h32[i] = v;
    h16[i] = (_Float16)v;
  }
#endif
  __syncthreads();

  const int lane = threadIdx.x & 31;
  const int j    = threadIdx.x >> 5;  // wave id == gate-column tile, 0..15
  const int nloc = lane & 15;
  const int mhi  = (lane >> 4) << 3;
  const int u    = j * 16 + nloc;
  const float brz = br[u], brr = br[Un + u], brh = br[2 * Un + u];
  const _Float16* WuZ = Wu + (size_t)(( 0 + j) * 16) * Un;
  const _Float16* WuR = Wu + (size_t)((16 + j) * 16) * Un;
  const _Float16* WuH = Wu + (size_t)((32 + j) * 16) * Un;

  for (int s = 0; s < Tn; ++s) {
    const int t_eff = dir ? (Tn - 1 - s) : s;
    v8f az[4] = {}, ar[4] = {}, ah[4] = {};

    for (int kt = 0; kt < Un; kt += 32) {
      v16h bz = load_b_h(WuZ + kt, Un, lane);
      v16h bb = load_b_h(WuR + kt, Un, lane);
      v16h bh = load_b_h(WuH + kt, Un, lane);
#pragma unroll
      for (int mi = 0; mi < 4; ++mi) {
        v16h a = load_a_h(&h16[(size_t)(mi * 16) * Un + kt], Un, lane);
        az[mi] = wmma16(a, bz, az[mi]);
        ar[mi] = wmma16(a, bb, ar[mi]);
        ah[mi] = wmma16(a, bh, ah[mi]);
      }
    }
    __syncthreads();  // all reads of h done before updates

#pragma unroll
    for (int mi = 0; mi < 4; ++mi) {
#pragma unroll
      for (int r = 0; r < 8; ++r) {
        const int b = mi * 16 + mhi + r;
        const size_t grow = ((size_t)b * Tn + t_eff) * G3;
        const float xz = (float)gx[grow + u];
        const float xr = (float)gx[grow + Un + u];
        const float xh = (float)gx[grow + 2 * Un + u];
        const float z  = sigmf(xz + az[mi][r] + brz);
        const float rr = sigmf(xr + ar[mi][r] + brr);
        const float hh = tanhf(xh + rr * (ah[mi][r] + brh));
        const float hold = h32[(size_t)b * Un + u];
        const float hnew = z * hold + (1.0f - z) * hh;
        h32[(size_t)b * Un + u] = hnew;
        h16[(size_t)b * Un + u] = (_Float16)hnew;
        if (seq_out)
          seq_out[((size_t)b * Tn + s) * (2 * Un) + seq_col + u] = (_Float16)hnew;
      }
    }
    __syncthreads();  // updates visible before next step's reads
  }

  if (hid_out) {
    for (int i = threadIdx.x; i < Bn * Un; i += blockDim.x) {
      const int b = i / Un, uu = i % Un;
      hid_out[(size_t)b * (2 * Un) + hid_col + uu] = h16[i];
    }
  }
}

// ---------------------------------------------------------------------------
// Head: out[64,256] = tanh(H[64,512](f16) @ dwT[256,512]^T + db). 1 block.
// ---------------------------------------------------------------------------
__global__ __launch_bounds__(256) void k_head(
    const _Float16* __restrict__ H, const _Float16* __restrict__ WT,
    const float* __restrict__ db, float* __restrict__ out) {
  const int lane = threadIdx.x & 31;
  const int wave = threadIdx.x >> 5;
  const int nloc = lane & 15;
  const int mhi  = (lane >> 4) << 3;
  for (int tile = wave; tile < 64; tile += 8) {
    const int mi = tile >> 4;  // 0..3
    const int ni = tile & 15;  // 0..15
    v8f acc = {};
    for (int kt = 0; kt < 2 * Un; kt += 32) {
      v16h a = load_a_h(H + (size_t)(mi * 16) * (2 * Un) + kt, 2 * Un, lane);
      v16h b = load_b_h(WT + (size_t)(ni * 16) * (2 * Un) + kt, 2 * Un, lane);
      acc = wmma16(a, b, acc);
    }
    const int n = ni * 16 + nloc;
    const float bv = db[n];
#pragma unroll
    for (int r = 0; r < 8; ++r) {
      const int m = mi * 16 + mhi + r;
      out[(size_t)m * Un + n] = tanhf(acc[r] + bv);
    }
  }
}

// ---------------------------------------------------------------------------
extern "C" void kernel_launch(void* const* d_in, const int* in_sizes, int n_in,
                              void* d_out, int out_size, void* d_ws, size_t ws_size,
                              hipStream_t stream) {
  (void)in_sizes; (void)n_in; (void)out_size; (void)ws_size;
  const float* h0  = (const float*)d_in[0];
  const float* x   = (const float*)d_in[1];
  const float* k1f = (const float*)d_in[2];
  const float* u1f = (const float*)d_in[3];
  const float* b1f = (const float*)d_in[4];
  const float* k1b = (const float*)d_in[5];
  const float* u1b = (const float*)d_in[6];
  const float* b1b = (const float*)d_in[7];
  const float* k2f = (const float*)d_in[8];
  const float* u2f = (const float*)d_in[9];
  const float* b2f = (const float*)d_in[10];
  const float* k2b = (const float*)d_in[11];
  const float* u2b = (const float*)d_in[12];
  const float* b2b = (const float*)d_in[13];
  const float* dw  = (const float*)d_in[14];
  const float* db  = (const float*)d_in[15];
  float* out = (float*)d_out;

  char* ws = (char*)d_ws;
  size_t off = 0;
  auto alloc = [&](size_t bytes) {
    char* p = ws + off;
    off += (bytes + 255) & ~(size_t)255;
    return p;
  };
  const size_t BT = (size_t)Bn * Tn;
  _Float16* x16    = (_Float16*)alloc(BT * Dn * 2);             // 16.8 MB
  _Float16* wk1f   = (_Float16*)alloc((size_t)G3 * Dn * 2);     // [768,256]
  _Float16* wk1b   = (_Float16*)alloc((size_t)G3 * Dn * 2);
  _Float16* wu1f   = (_Float16*)alloc((size_t)G3 * Un * 2);
  _Float16* wu1b   = (_Float16*)alloc((size_t)G3 * Un * 2);
  _Float16* wk2f   = (_Float16*)alloc((size_t)G3 * 2 * Un * 2); // [768,512]
  _Float16* wk2b   = (_Float16*)alloc((size_t)G3 * 2 * Un * 2);
  _Float16* wu2f   = (_Float16*)alloc((size_t)G3 * Un * 2);
  _Float16* wu2b   = (_Float16*)alloc((size_t)G3 * Un * 2);
  _Float16* wdw    = (_Float16*)alloc((size_t)Un * 2 * Un * 2); // [256,512]
  _Float16* gx1f   = (_Float16*)alloc(BT * G3 * 2);             // 50.3 MB each
  _Float16* gx1b   = (_Float16*)alloc(BT * G3 * 2);
  _Float16* gx2f   = (_Float16*)alloc(BT * G3 * 2);
  _Float16* gx2b   = (_Float16*)alloc(BT * G3 * 2);
  _Float16* pred16 = (_Float16*)alloc(BT * 2 * Un * 2);         // 33.5 MB
  _Float16* hid16  = (_Float16*)alloc((size_t)Bn * 2 * Un * 2);

  const int TB = 256;
  // ---- weight / input conversion ----
  {
    int n = (int)(BT * Dn);
    k_cvt_f16<<<(n + TB - 1) / TB, TB, 0, stream>>>(x, x16, n);
  }
  auto tr = [&](const float* src, _Float16* dst, int K, int N) {
    int n = K * N;
    k_transpose_f16<<<(n + TB - 1) / TB, TB, 0, stream>>>(src, dst, K, N);
  };
  tr(k1f, wk1f, Dn, G3);      tr(k1b, wk1b, Dn, G3);
  tr(u1f, wu1f, Un, G3);      tr(u1b, wu1b, Un, G3);
  tr(k2f, wk2f, 2 * Un, G3);  tr(k2b, wk2b, 2 * Un, G3);
  tr(u2f, wu2f, Un, G3);      tr(u2b, wu2b, Un, G3);
  tr(dw,  wdw,  2 * Un, Un);

  // ---- layer 1 input projections: gx = x @ Wk + bi ----
  dim3 g1(G3 / 64, (int)(BT / 256));
  k_gemm_bias<<<g1, TB, 0, stream>>>(x16, wk1f, b1f, gx1f, (int)BT, G3, Dn);
  k_gemm_bias<<<g1, TB, 0, stream>>>(x16, wk1b, b1b, gx1b, (int)BT, G3, Dn);

  // ---- layer 1 scan (fwd + bwd), writes pred16 [B,T,512] ----
  k_gru_scan<<<2, 512, 0, stream>>>(gx1f, gx1b, wu1f, wu1b,
                                    b1f + G3, b1b + G3, h0,
                                    pred16, 0, Un, nullptr, 0, 0);

  // ---- layer 2 input projections: gx2 = pred @ Wk2 + bi ----
  k_gemm_bias<<<g1, TB, 0, stream>>>(pred16, wk2f, b2f, gx2f, (int)BT, G3, 2 * Un);
  k_gemm_bias<<<g1, TB, 0, stream>>>(pred16, wk2b, b2b, gx2b, (int)BT, G3, 2 * Un);

  // ---- layer 2 scan, final hidden only -> hid16 [64,512] ----
  k_gru_scan<<<2, 512, 0, stream>>>(gx2f, gx2b, wu2f, wu2b,
                                    b2f + G3, b2b + G3, h0,
                                    nullptr, 0, 0, hid16, 0, Un);

  // ---- head: tanh(hidden @ dw + db) ----
  k_head<<<1, TB, 0, stream>>>(hid16, wdw, db, out);
}
